// Attention_lora_8169027797027
// MI455X (gfx1250) — matrix operand order repaired and checked by reading, not compiled, not run on hardware
//
#include <hip/hip_runtime.h>

typedef __bf16 bh;
typedef __attribute__((ext_vector_type(16))) __bf16 v16bh;
typedef __attribute__((ext_vector_type(8)))  __bf16 v8bh;
typedef __attribute__((ext_vector_type(8)))  float  v8f;
typedef __attribute__((ext_vector_type(4)))  int    v4i;

typedef v4i __attribute__((address_space(1))) as1_v4i;
typedef v4i __attribute__((address_space(3))) as3_v4i;

#if defined(__has_builtin)
#if __has_builtin(__builtin_amdgcn_global_load_async_to_lds_b128) && \
    __has_builtin(__builtin_amdgcn_s_wait_asynccnt)
#define HAVE_ASYNC_LDS 1
#endif
#endif
#ifndef HAVE_ASYNC_LDS
#define HAVE_ASYNC_LDS 0
#endif

union ABfrag { v16bh v; v8bh h[2]; };

static __device__ __forceinline__ v8f wmma_bf16(v16bh a, v16bh b, v8f c) {
  return __builtin_amdgcn_wmma_f32_16x16x32_bf16(false, a, false, b, (short)0, c,
                                                 false, false);
}

constexpr int kB = 8, kN = 1024, kC = 768, kH = 12, kHD = 64, kR = 16;
constexpr int kM = kB * kN;        // 8192 token rows
constexpr float kScale = 0.125f;   // 64^-0.5

// ---------------- conversion / layout kernels ----------------
__global__ void cast_f32_to_bf16(const float* __restrict__ src,
                                 bh* __restrict__ dst, int n) {
  int i = blockIdx.x * blockDim.x + threadIdx.x;
  if (i < n) dst[i] = (bh)src[i];
}

// Wt[n][k] = W[k][n]  (768 x 768)
__global__ void transpose_w_bf16(const float* __restrict__ W,
                                 bh* __restrict__ Wt) {
  int i = blockIdx.x * blockDim.x + threadIdx.x;
  if (i >= kC * kC) return;
  int n = i / kC, k = i % kC;
  Wt[n * kC + k] = (bh)W[k * kC + n];
}

// laT[r][k] = la[k][r]  (16 x 768)
__global__ void transpose_la_bf16(const float* __restrict__ la,
                                  bh* __restrict__ laT) {
  int i = blockIdx.x * blockDim.x + threadIdx.x;
  if (i >= kR * kC) return;
  int r = i / kC, k = i % kC;
  laT[r * kC + k] = (bh)la[k * kR + r];
}

// lbT[n][r] = lb[r][n] for r<16, zero pad r in [16,32)  (768 x 32)
__global__ void transpose_lb_bf16(const float* __restrict__ lb,
                                  bh* __restrict__ lbT) {
  int i = blockIdx.x * blockDim.x + threadIdx.x;
  if (i >= kC * 32) return;
  int n = i / 32, r = i % 32;
  lbT[n * 32 + r] = (r < kR) ? (bh)lb[r * kC + n] : (bh)0.0f;
}

// ---------------- LoRA mid: t = bw * (x @ la), bf16 [3][8192][32] padded ----
__global__ void lora_mid_kernel(const bh* __restrict__ xb,
                                const bh* __restrict__ laT,    // [3][16][768]
                                const float* __restrict__ bw,  // [3]
                                bh* __restrict__ tmid) {       // [3][8192][32]
  int lane = threadIdx.x;
  int mt = blockIdx.x;        // 512 row tiles
  int proj = blockIdx.y;      // 3
  int half = lane >> 4;
  int l15 = lane & 15;
  int arow = mt * 16 + l15;
  int ak0 = half ? 16 : 0;
  const bh* laTp = laT + proj * (kR * kC);
  v8f acc = {};
  for (int kk = 0; kk < kC; kk += 32) {
    ABfrag a;
    a.h[0] = *(const v8bh*)(xb + arow * kC + kk + ak0);
    a.h[1] = *(const v8bh*)(xb + arow * kC + kk + ak0 + 8);
    v16bh b = *(const v16bh*)(laTp + l15 * kC + kk + (half ? 16 : 0));
    acc = wmma_bf16(a.v, b, acc);
  }
  float s = bw[proj];
  bh* tp = tmid + proj * (kM * 32);
#pragma unroll
  for (int r = 0; r < 8; ++r) {
    int m = mt * 16 + r + half * 8;
    tp[m * 32 + l15] = (bh)(acc[r] * s);
    tp[m * 32 + 16 + l15] = (bh)0.0f;  // zero K-pad for A-fragment reuse
  }
}

// ---------------- fused QKV projection + LoRA update ----------------
__global__ void qkv_gemm_kernel(const bh* __restrict__ xb,    // [8192][768]
                                const bh* __restrict__ Wt,    // [3][768][768]
                                const bh* __restrict__ lbT,   // [3][768][32]
                                const bh* __restrict__ tmid,  // [3][8192][32]
                                bh* __restrict__ qh,          // [B,H,N,HD]
                                bh* __restrict__ kh,          // [B,H,N,HD]
                                bh* __restrict__ vT) {        // [B,H,HD,N]
  int lane = threadIdx.x;
  int tile = blockIdx.x;      // (8192/16)*(768/64) = 6144
  int proj = blockIdx.y;      // 0=q 1=k 2=v
  int mt = tile / (kC / 64);
  int nt = tile % (kC / 64);
  int half = lane >> 4;
  int l15 = lane & 15;
  int arow = mt * 16 + l15;
  int ak0 = half ? 16 : 0;
  const bh* Wp = Wt + (size_t)proj * kC * kC;
  v8f acc[4] = {{}, {}, {}, {}};
  for (int kk = 0; kk < kC; kk += 32) {
    __builtin_prefetch(xb + arow * kC + kk + 64, 0, 1);
    ABfrag a;
    a.h[0] = *(const v8bh*)(xb + arow * kC + kk + ak0);
    a.h[1] = *(const v8bh*)(xb + arow * kC + kk + ak0 + 8);
    int bo = kk + (half ? 16 : 0);
#pragma unroll
    for (int j = 0; j < 4; ++j) {
      v16bh b = *(const v16bh*)(Wp + (nt * 64 + j * 16 + l15) * kC + bo);
      acc[j] = wmma_bf16(a.v, b, acc[j]);
    }
  }
  {  // rank-16 LoRA update, one padded K=32 WMMA per column tile
    ABfrag a;
    const bh* tp = tmid + proj * (kM * 32);
    a.h[0] = *(const v8bh*)(tp + arow * 32 + ak0);
    a.h[1] = *(const v8bh*)(tp + arow * 32 + ak0 + 8);
    const bh* lp = lbT + proj * (kC * 32);
    int bo = half ? 16 : 0;
#pragma unroll
    for (int j = 0; j < 4; ++j) {
      v16bh b = *(const v16bh*)(lp + (nt * 64 + j * 16 + l15) * 32 + bo);
      acc[j] = wmma_bf16(a.v, b, acc[j]);
    }
  }
#pragma unroll
  for (int j = 0; j < 4; ++j) {
    int col = nt * 64 + j * 16 + l15;
    int h = col >> 6;
    int d = col & 63;
#pragma unroll
    for (int r = 0; r < 8; ++r) {
      int grow = mt * 16 + r + half * 8;
      int b = grow >> 10;
      int ntok = grow & 1023;
      bh val = (bh)acc[j][r];
      if (proj == 0)
        qh[((size_t)(b * kH + h) * kN + ntok) * kHD + d] = val;
      else if (proj == 1)
        kh[((size_t)(b * kH + h) * kN + ntok) * kHD + d] = val;
      else
        vT[((size_t)(b * kH + h) * kHD + d) * kN + ntok] = val;
    }
  }
}

// ---------------- attention: one wave per (b, h, 16-query tile) -----------
// LDS: [0,64K)   S (f32 16x1024) overlaid by P (bf16 16x1024)
//      [64K,68K) K-tile double buffer (2 x 2KB)
//      [68K,76K) V-tile double buffer (2 x 4KB)
__global__ void attention_kernel(const bh* __restrict__ qh,
                                 const bh* __restrict__ kh,
                                 const bh* __restrict__ vT,
                                 bh* __restrict__ attnO) {  // [8192][768]
  extern __shared__ char smem[];
  float* S = (float*)smem;  // [16][1024] f32 scores
  bh* P = (bh*)smem;        // [16][1024] bf16 probs, overlays consumed S rows
  bh* Kst = (bh*)(smem + 65536);
  bh* Vst = (bh*)(smem + 65536 + 4096);
  int lane = threadIdx.x;
  int idx = blockIdx.x;  // B*H*(N/16) = 6144
  int qt = idx & 63;
  int h = (idx >> 6) % kH;
  int b = idx / (64 * kH);
  int half = lane >> 4;
  int l15 = lane & 15;
  int ak0 = half ? 16 : 0;
  int bo = half ? 16 : 0;
  const bh* Qp = qh + (size_t)(b * kH + h) * kN * kHD;
  const bh* Kp = kh + (size_t)(b * kH + h) * kN * kHD;
  const bh* Vp = vT + (size_t)(b * kH + h) * kHD * kN;

  // Q A-fragments: 16 rows x 64, two K=32 slices
  ABfrag qa0, qa1;
  {
    const bh* qrow = Qp + (qt * 16 + l15) * kHD;
    qa0.h[0] = *(const v8bh*)(qrow + ak0);
    qa0.h[1] = *(const v8bh*)(qrow + ak0 + 8);
    qa1.h[0] = *(const v8bh*)(qrow + 32 + ak0);
    qa1.h[1] = *(const v8bh*)(qrow + 32 + ak0 + 8);
  }

  // ---- S = (Q K^T) * scale -> LDS, K tiles double-buffered via async LDS ----
#if HAVE_ASYNC_LDS
  {
    // K tile kt is 16 rows x 64 bf16 = 2048B, contiguous in memory.
    auto issueK = [&](int kt, bh* dst) {
      const bh* g = Kp + kt * 16 * kHD;
      for (int c = lane; c < 128; c += 32)  // 128 x 16B chunks
        __builtin_amdgcn_global_load_async_to_lds_b128(
            (as1_v4i*)(g + c * 8), (as3_v4i*)(dst + c * 8), 0, 0);
    };
    issueK(0, Kst);
    for (int kt = 0; kt < kN / 16; ++kt) {
      if (kt + 1 < kN / 16) {
        issueK(kt + 1, Kst + ((kt + 1) & 1) * 1024);
        __builtin_amdgcn_s_wait_asynccnt(4);  // current tile's 4 copies done
      } else {
        __builtin_amdgcn_s_wait_asynccnt(0);
      }
      asm volatile("" ::: "memory");
      const bh* Kt = Kst + (kt & 1) * 1024;
      v16bh b0 = *(const v16bh*)(Kt + l15 * kHD + bo);
      v16bh b1 = *(const v16bh*)(Kt + l15 * kHD + 32 + bo);
      v8f s = {};
      s = wmma_bf16(qa0.v, b0, s);
      s = wmma_bf16(qa1.v, b1, s);
#pragma unroll
      for (int r = 0; r < 8; ++r)
        S[(r + half * 8) * kN + kt * 16 + l15] = s[r] * kScale;
    }
  }
#else
  for (int kt = 0; kt < kN / 16; ++kt) {
    int key = kt * 16 + l15;
    __builtin_prefetch(Kp + (key + 16) * kHD, 0, 1);
    v16bh b0 = *(const v16bh*)(Kp + key * kHD + bo);
    v16bh b1 = *(const v16bh*)(Kp + key * kHD + 32 + bo);
    v8f s = {};
    s = wmma_bf16(qa0.v, b0, s);
    s = wmma_bf16(qa1.v, b1, s);
#pragma unroll
    for (int r = 0; r < 8; ++r)
      S[(r + half * 8) * kN + kt * 16 + l15] = s[r] * kScale;
  }
#endif
  __syncthreads();

  // ---- row softmax (f32), write bf16 probabilities in place ----
  for (int m = 0; m < 16; ++m) {
    float mx = -3.0e38f;
    for (int i = lane; i < kN; i += 32) mx = fmaxf(mx, S[m * kN + i]);
    for (int off = 16; off > 0; off >>= 1) mx = fmaxf(mx, __shfl_xor(mx, off, 32));
    float sum = 0.f;
    for (int i = lane; i < kN; i += 32) {
      float e = __expf(S[m * kN + i] - mx);
      S[m * kN + i] = e;
      sum += e;
    }
    for (int off = 16; off > 0; off >>= 1) sum += __shfl_xor(sum, off, 32);
    float inv = 1.0f / sum;
    for (int i = lane; i < kN; i += 32)
      P[m * kN + i] = (bh)(S[m * kN + i] * inv);  // clobbers only S rows <= m
  }
  __syncthreads();

  // ---- O = P @ V, V tiles (64 x 32 bf16 = 4KB) double-buffered ----
  v8f o[4] = {{}, {}, {}, {}};
#if HAVE_ASYNC_LDS
  {
    auto issueV = [&](int mt2, bh* dst) {
      for (int c = lane; c < 256; c += 32) {  // 256 x 16B chunks
        int d = c >> 2, u = c & 3;
        __builtin_amdgcn_global_load_async_to_lds_b128(
            (as1_v4i*)(Vp + d * kN + mt2 * 32 + u * 8),
            (as3_v4i*)(dst + c * 8), 0, 0);
      }
    };
    issueV(0, Vst);
    for (int mt2 = 0; mt2 < kN / 32; ++mt2) {
      if (mt2 + 1 < kN / 32) {
        issueV(mt2 + 1, Vst + ((mt2 + 1) & 1) * 2048);
        __builtin_amdgcn_s_wait_asynccnt(8);  // current tile's 8 copies done
      } else {
        __builtin_amdgcn_s_wait_asynccnt(0);
      }
      asm volatile("" ::: "memory");
      const bh* Vt = Vst + (mt2 & 1) * 2048;
      ABfrag pa;
      pa.h[0] = *(const v8bh*)(P + l15 * kN + mt2 * 32 + ak0);
      pa.h[1] = *(const v8bh*)(P + l15 * kN + mt2 * 32 + ak0 + 8);
#pragma unroll
      for (int j = 0; j < 4; ++j) {
        v16bh vb = *(const v16bh*)(Vt + (j * 16 + l15) * 32 + bo);
        o[j] = wmma_bf16(pa.v, vb, o[j]);
      }
    }
  }
#else
  for (int mt2 = 0; mt2 < kN / 32; ++mt2) {
    ABfrag pa;
    pa.h[0] = *(const v8bh*)(P + l15 * kN + mt2 * 32 + ak0);
    pa.h[1] = *(const v8bh*)(P + l15 * kN + mt2 * 32 + ak0 + 8);
    int vbo = mt2 * 32 + bo;
#pragma unroll
    for (int j = 0; j < 4; ++j) {
      v16bh vb = *(const v16bh*)(Vp + (j * 16 + l15) * kN + vbo);
      o[j] = wmma_bf16(pa.v, vb, o[j]);
    }
  }
#endif
#pragma unroll
  for (int j = 0; j < 4; ++j) {
    int d = j * 16 + l15;
#pragma unroll
    for (int r = 0; r < 8; ++r) {
      int qrow = qt * 16 + r + half * 8;
      attnO[(size_t)(b * kN + qrow) * kC + h * kHD + d] = (bh)o[j][r];
    }
  }
}

// ---------------- output projection + bias (f32 out) ----------------
__global__ void proj_gemm_kernel(const bh* __restrict__ aO,   // [8192][768]
                                 const bh* __restrict__ WtP,  // [768][768]
                                 const float* __restrict__ bp,
                                 float* __restrict__ out) {   // [8192][768]
  int lane = threadIdx.x;
  int tile = blockIdx.x;
  int mt = tile / (kC / 64);
  int nt = tile % (kC / 64);
  int half = lane >> 4;
  int l15 = lane & 15;
  int arow = mt * 16 + l15;
  int ak0 = half ? 16 : 0;
  v8f acc[4] = {{}, {}, {}, {}};
  for (int kk = 0; kk < kC; kk += 32) {
    __builtin_prefetch(aO + arow * kC + kk + 64, 0, 1);
    ABfrag a;
    a.h[0] = *(const v8bh*)(aO + arow * kC + kk + ak0);
    a.h[1] = *(const v8bh*)(aO + arow * kC + kk + ak0 + 8);
    int bo = kk + (half ? 16 : 0);
#pragma unroll
    for (int j = 0; j < 4; ++j) {
      v16bh b = *(const v16bh*)(WtP + (nt * 64 + j * 16 + l15) * kC + bo);
      acc[j] = wmma_bf16(a.v, b, acc[j]);
    }
  }
#pragma unroll
  for (int j = 0; j < 4; ++j) {
    int col = nt * 64 + j * 16 + l15;
    float bias = bp[col];
#pragma unroll
    for (int r = 0; r < 8; ++r) {
      int grow = mt * 16 + r + half * 8;
      out[(size_t)grow * kC + col] = acc[j][r] + bias;
    }
  }
}

extern "C" void kernel_launch(void* const* d_in, const int* in_sizes, int n_in,
                              void* d_out, int out_size, void* d_ws,
                              size_t ws_size, hipStream_t stream) {
  (void)in_sizes; (void)n_in; (void)out_size; (void)ws_size;
  const float* x    = (const float*)d_in[0];
  const float* Wq   = (const float*)d_in[1];
  const float* Wk   = (const float*)d_in[2];
  const float* Wv   = (const float*)d_in[3];
  const float* Wp   = (const float*)d_in[4];
  const float* bp   = (const float*)d_in[5];
  const float* la_q = (const float*)d_in[6];
  const float* lb_q = (const float*)d_in[7];
  const float* la_k = (const float*)d_in[8];
  const float* lb_k = (const float*)d_in[9];
  const float* la_v = (const float*)d_in[10];
  const float* lb_v = (const float*)d_in[11];
  const float* bw   = (const float*)d_in[12];

  char* ws = (char*)d_ws;
  size_t off = 0;
  auto take = [&](size_t bytes) -> char* {
    char* p = ws + off;
    off += (bytes + 255) & ~(size_t)255;
    return p;
  };
  bh* xb    = (bh*)take((size_t)kM * kC * 2);      // 12.6 MB
  bh* Wt    = (bh*)take((size_t)4 * kC * kC * 2);  // 4.7 MB (q,k,v,p)
  bh* laT   = (bh*)take((size_t)3 * kR * kC * 2);
  bh* lbT   = (bh*)take((size_t)3 * kC * 32 * 2);
  bh* tmid  = (bh*)take((size_t)3 * kM * 32 * 2);
  bh* qhB   = (bh*)take((size_t)kM * kC * 2);
  bh* khB   = (bh*)take((size_t)kM * kC * 2);
  bh* vTB   = (bh*)take((size_t)kM * kC * 2);
  bh* attnO = (bh*)take((size_t)kM * kC * 2);

  int n_x = kM * kC;
  cast_f32_to_bf16<<<(n_x + 255) / 256, 256, 0, stream>>>(x, xb, n_x);
  int n_w = kC * kC;
  transpose_w_bf16<<<(n_w + 255) / 256, 256, 0, stream>>>(Wq, Wt + 0 * (size_t)n_w);
  transpose_w_bf16<<<(n_w + 255) / 256, 256, 0, stream>>>(Wk, Wt + 1 * (size_t)n_w);
  transpose_w_bf16<<<(n_w + 255) / 256, 256, 0, stream>>>(Wv, Wt + 2 * (size_t)n_w);
  transpose_w_bf16<<<(n_w + 255) / 256, 256, 0, stream>>>(Wp, Wt + 3 * (size_t)n_w);
  int n_la = kR * kC, n_lb = kC * 32;
  transpose_la_bf16<<<(n_la + 255) / 256, 256, 0, stream>>>(la_q, laT + 0 * n_la);
  transpose_la_bf16<<<(n_la + 255) / 256, 256, 0, stream>>>(la_k, laT + 1 * n_la);
  transpose_la_bf16<<<(n_la + 255) / 256, 256, 0, stream>>>(la_v, laT + 2 * n_la);
  transpose_lb_bf16<<<(n_lb + 255) / 256, 256, 0, stream>>>(lb_q, lbT + 0 * n_lb);
  transpose_lb_bf16<<<(n_lb + 255) / 256, 256, 0, stream>>>(lb_k, lbT + 1 * n_lb);
  transpose_lb_bf16<<<(n_lb + 255) / 256, 256, 0, stream>>>(lb_v, lbT + 2 * n_lb);

  lora_mid_kernel<<<dim3(kM / 16, 3), 32, 0, stream>>>(xb, laT, bw, tmid);
  qkv_gemm_kernel<<<dim3((kM / 16) * (kC / 64), 3), 32, 0, stream>>>(
      xb, Wt, lbT, tmid, qhB, khB, vTB);
  attention_kernel<<<kB * kH * (kN / 16), 32, 76 * 1024, stream>>>(qhB, khB, vTB,
                                                                   attnO);
  proj_gemm_kernel<<<(kM / 16) * (kC / 64), 32, 0, stream>>>(
      attnO, Wt + 3 * (size_t)n_w, bp, (float*)d_out);
}
